// ImageLanguageAttentionCombiner_9689446220212
// MI455X (gfx1250) — matrix operand order repaired and checked by reading, not compiled, not run on hardware
//
#include <hip/hip_runtime.h>
#include <hip/hip_bf16.h>

typedef __attribute__((ext_vector_type(16))) __bf16 v16bf;
typedef __attribute__((ext_vector_type(8)))  __bf16 v8bf;
typedef __attribute__((ext_vector_type(4)))  __bf16 v4bf;
typedef __attribute__((ext_vector_type(8)))  float  v8f;

#define B_   256
#define C_   256
#define NH_  8
#define L_   20
#define HW_  256
#define DH_  32   // q/k head dim
#define DV_  64   // v head dim
#define OUTB (C_ + 2 * C_ + HW_ * 2 * C_)   // 256 + 512 + 131072 = 131840 per batch

static __device__ inline float wave_max(float v) {
#pragma unroll
  for (int m = 16; m > 0; m >>= 1) v = fmaxf(v, __shfl_xor(v, m, 32));
  return v;
}
static __device__ inline float wave_sum(float v) {
#pragma unroll
  for (int m = 16; m > 0; m >>= 1) v += __shfl_xor(v, m, 32);
  return v;
}

// One wave per (batch, head).
__global__ __launch_bounds__(32)
void attn_kernel(const float* __restrict__ key_img,
                 const int*   __restrict__ qids,
                 const float* __restrict__ emb,
                 const float* __restrict__ scale_p,
                 float*       __restrict__ out)
{
  // Transposed logits buffer: [key][query]; rows q=20..31 are exact zeros
  // (zero-padded A rows), so no store guards are needed anywhere.
  __shared__ __bf16 s_lt[HW_][32];     // 16 KB : logits^T, later attn_w^T
  __shared__ __bf16 s_vh[HW_][DV_];    // 32 KB : staged V (bf16)
  __shared__ float  s_pt[DV_][32];     //  8 KB : (attn_w @ V)^T
  __shared__ float  s_sm[HW_];         //  1 KB : attn_softmax (raw-score path)

  const int b    = blockIdx.x >> 3;
  const int h    = blockIdx.x & 7;
  const int lane = threadIdx.x;

  // ISA lane-layout helpers (wave32, 16x16 WMMA tiles)
  const int rowq  = lane & 15;          // A: matrix row this lane owns
  const int kb    = (lane >> 4) * 8;    // unused; A loads use khalf (k = 16*half + e)
  const int ncol  = lane & 15;          // B/C/D: matrix column this lane owns
  const int khalf = (lane >> 4) * 16;   // B: K half per lane half
  const int rb    = (lane >> 4) * 8;    // C/D: M base per lane half

  // ---------------- Q fragments: two 16x32 bf16 A tiles (rows 20..31 zero) ---
  v16bf a0, a1;
  {
    const int id0 = qids[b * L_ + rowq];
    const float* q0p = emb + (size_t)id0 * C_ + h * DH_;
    const int   q1  = 16 + rowq;
    const float m1  = (q1 < L_) ? 1.0f : 0.0f;     // branchless zero-pad
    const int   q1c = (q1 < L_) ? q1 : (L_ - 1);
    const int id1 = qids[b * L_ + q1c];
    const float* q1p = emb + (size_t)id1 * C_ + h * DH_;
#pragma unroll
    for (int e = 0; e < 8; ++e) {
      a0[e]     = (__bf16)q0p[khalf + e];
      a0[e + 8] = (__bf16)q0p[khalf + 8 + e];
      a1[e]     = (__bf16)(m1 * q1p[khalf + e]);
      a1[e + 8] = (__bf16)(m1 * q1p[khalf + 8 + e]);
    }
  }

  // ---------------- logits = Q K^T over 16 key tiles --------------------------
  for (int t = 0; t < 16; ++t) {
    v16bf bk;  // B tile 32x16: column = key (per lane), K = channel
    const int key = t * 16 + ncol;
    const float* kp = key_img + ((size_t)b * HW_ + key) * C_ + h * DH_ + khalf;
#pragma unroll
    for (int e = 0; e < 16; ++e) bk[e] = (__bf16)kp[e];

    v8f c0 = {}; v8f c1 = {};
    c0 = __builtin_amdgcn_wmma_f32_16x16x32_bf16(false, a0, false, bk, (short)0, c0, false, false);
    c1 = __builtin_amdgcn_wmma_f32_16x16x32_bf16(false, a1, false, bk, (short)0, c1, false, false);

    // D layout: this lane owns column `key`, rows rb..rb+7 (c0) / 16+rb.. (c1).
    v8bf w0, w1;
#pragma unroll
    for (int r = 0; r < 8; ++r) { w0[r] = (__bf16)c0[r]; w1[r] = (__bf16)c1[r]; }
    *(v8bf*)&s_lt[key][rb]      = w0;   // ds_store_b128
    *(v8bf*)&s_lt[key][16 + rb] = w1;   // ds_store_b128
  }
  __syncthreads();

  // ------------- raw-score path: mean over q, softmax over keys ---------------
  {
    float sc[8]; float lmax = -3.4e38f;
#pragma unroll
    for (int i = 0; i < 8; ++i) {
      const int col = lane + 32 * i;
      float s = 0.f;
#pragma unroll
      for (int q = 0; q < L_; ++q) s += (float)s_lt[col][q];
      sc[i] = s * (1.0f / L_);
      lmax = fmaxf(lmax, sc[i]);
    }
    const float gmax = wave_max(lmax);
    float lsum = 0.f;
#pragma unroll
    for (int i = 0; i < 8; ++i) { sc[i] = expf(sc[i] - gmax); lsum += sc[i]; }
    const float inv = 1.0f / wave_sum(lsum);
#pragma unroll
    for (int i = 0; i < 8; ++i) s_sm[lane + 32 * i] = sc[i] * inv;
  }
  __syncthreads();

  // ------------- residual path: stream attn_softmax * vh; stage vh bf16 -------
  // 2 keys x 16 lanes x float4 per iteration -> b128 global stores.
  {
    const int dk = lane >> 4;          // which of the 2 keys this lane handles
    const int d0 = (lane & 15) * 4;    // 4 consecutive v-channels
    float* obase = out + (size_t)b * OUTB + 3 * C_ + h * DV_ + d0;
    for (int kk = 0; kk < HW_; kk += 2) {
      const int key = kk + dk;
      const float wgt = s_sm[key];
      float4 v;
      if (h < 4) {
        v = *(const float4*)(key_img + ((size_t)b * HW_ + key) * C_ + h * DV_ + d0);
      } else {
        // positional half: channel parity selects x=(k%16)/16 vs y=(k/16)/16
        const float x = (float)(key & 15) * 0.0625f;
        const float y = (float)(key >> 4) * 0.0625f;
        v = make_float4(x, y, x, y);
      }
      v4bf vb;
      vb[0] = (__bf16)v.x; vb[1] = (__bf16)v.y;
      vb[2] = (__bf16)v.z; vb[3] = (__bf16)v.w;
      *(v4bf*)&s_vh[key][d0] = vb;     // ds_store_b64
      float4 o = make_float4(wgt * v.x, wgt * v.y, wgt * v.z, wgt * v.w);
      *(float4*)(obase + (size_t)key * (2 * C_)) = o;   // global_store_b128
    }
  }
  __syncthreads();

  // ------------- scaled per-row softmax -> attn_w (overwrite s_lt) ------------
  const float scale = scale_p[0];
  for (int q = 0; q < L_; ++q) {
    float v[8]; float lmax = -3.4e38f;
#pragma unroll
    for (int i = 0; i < 8; ++i) {
      v[i] = scale * (float)s_lt[lane + 32 * i][q];
      lmax = fmaxf(lmax, v[i]);
    }
    const float gmax = wave_max(lmax);
    float lsum = 0.f;
#pragma unroll
    for (int i = 0; i < 8; ++i) { v[i] = expf(v[i] - gmax); lsum += v[i]; }
    const float inv = 1.0f / wave_sum(lsum);
#pragma unroll
    for (int i = 0; i < 8; ++i) s_lt[lane + 32 * i][q] = (__bf16)(v[i] * inv);
  }
  __syncthreads();

  // ------------- P = attn_w @ vh via WMMA (K=256 in 8 chunks, N=64 in 4) ------
  // Rows q=20..31 of s_lt stayed zero, so no A-fragment guards needed.
  v8f acc[2][4];
  {
    v8f z = {};
#pragma unroll
    for (int m = 0; m < 2; ++m)
#pragma unroll
      for (int dt = 0; dt < 4; ++dt) acc[m][dt] = z;
  }
  for (int kc = 0; kc < 8; ++kc) {
    const int kcb = kc * 32;
    v16bf wa0, wa1;
#pragma unroll
    for (int e = 0; e < 8; ++e) {
      wa0[e]     = s_lt[kcb + khalf + e][rowq];
      wa0[e + 8] = s_lt[kcb + khalf + 8 + e][rowq];
      wa1[e]     = s_lt[kcb + khalf + e][16 + rowq];
      wa1[e + 8] = s_lt[kcb + khalf + 8 + e][16 + rowq];
    }
#pragma unroll
    for (int dt = 0; dt < 4; ++dt) {
      v16bf vb;
#pragma unroll
      for (int e = 0; e < 16; ++e) vb[e] = s_vh[kcb + khalf + e][dt * 16 + ncol];
      acc[0][dt] = __builtin_amdgcn_wmma_f32_16x16x32_bf16(false, wa0, false, vb, (short)0, acc[0][dt], false, false);
      acc[1][dt] = __builtin_amdgcn_wmma_f32_16x16x32_bf16(false, wa1, false, vb, (short)0, acc[1][dt], false, false);
    }
  }
  // D spill, transposed [d][q]: 8 consecutive q per lane -> wide ds stores.
#pragma unroll
  for (int m = 0; m < 2; ++m)
#pragma unroll
    for (int dt = 0; dt < 4; ++dt)
      *(v8f*)&s_pt[dt * 16 + ncol][m * 16 + rb] = acc[m][dt];
  __syncthreads();

  // ------------- pos_attention = mean over L of P -----------------------------
  {
    const int d0 = lane * 2;
    float s0 = 0.f, s1 = 0.f;
#pragma unroll
    for (int q = 0; q < L_; ++q) { s0 += s_pt[d0][q]; s1 += s_pt[d0 + 1][q]; }
    float* op = out + (size_t)b * OUTB + C_ + h * DV_ + d0;
    op[0] = s0 * (1.0f / L_);
    op[1] = s1 * (1.0f / L_);
  }
}

// query_encoding[b, c] = mean over L of emb_table[query_ids[b, q]][c]
__global__ __launch_bounds__(256)
void qenc_kernel(const int* __restrict__ qids,
                 const float* __restrict__ emb,
                 float* __restrict__ out)
{
  const int b = blockIdx.x;
  const int c = threadIdx.x;
  float s = 0.f;
  for (int q = 0; q < L_; ++q)
    s += emb[(size_t)qids[b * L_ + q] * C_ + c];
  out[(size_t)b * OUTB + c] = s * (1.0f / L_);
}

extern "C" void kernel_launch(void* const* d_in, const int* in_sizes, int n_in,
                              void* d_out, int out_size, void* d_ws, size_t ws_size,
                              hipStream_t stream) {
  const float* key_img = (const float*)d_in[0];
  const int*   qids    = (const int*)d_in[1];
  const float* emb     = (const float*)d_in[2];
  const float* scale   = (const float*)d_in[3];
  float* out = (float*)d_out;

  qenc_kernel<<<B_, C_, 0, stream>>>(qids, emb, out);
  attn_kernel<<<B_ * NH_, 32, 0, stream>>>(key_img, qids, emb, scale, out);
}
